// GATRegressor_16372415332827
// MI455X (gfx1250) — matrix operand order repaired and checked by reading, not compiled, not run on hardware
//
#include <hip/hip_runtime.h>
#include <math.h>

// GAT regressor for MI455X (gfx1250, wave32, WMMA).
// N=50000 nodes, E=800000 edges + N self loops.
#define NN 50000
#define NE 800000
#define NEG_SLOPE 0.2f

typedef __attribute__((ext_vector_type(16))) _Float16 v16h;
typedef __attribute__((ext_vector_type(8)))  float    v8f;

// ---------- order-preserving float <-> u32 encoding for atomic max ----------
static __device__ __forceinline__ unsigned encf(float f) {
  int b = __float_as_int(f);
  return (unsigned)(b >= 0 ? (b | 0x80000000) : ~b);
}
static __device__ __forceinline__ float decf(unsigned u) {
  int b = (u & 0x80000000u) ? (int)(u & 0x7fffffffu) : ~(int)u;
  return __int_as_float(b);
}

// ---------- pack f32 weight matrix W[K,N] into f16 WMMA B-operand layout ----
// Per (ntile, kstep): 32 lanes x 16 halves, lane holds column n = ntile*16 +
// (lane&15), halves 0..7 = K kbase..kbase+7, halves 8..15 = K kbase+16..kbase+23,
// kbase = kstep*32 + (lane<16?0:8).
__global__ void gat_pack_b(const float* __restrict__ W, _Float16* __restrict__ P,
                           int K, int N) {
  int idx = blockIdx.x * blockDim.x + threadIdx.x;
  if (idx >= K * N) return;
  int i    = idx & 15;
  int lane = (idx >> 4) & 31;
  int t    = idx >> 9;                 // ntile*ksteps + kstep
  int ksteps = K >> 5;
  int ntile = t / ksteps;
  int kstep = t - ntile * ksteps;
  int k = kstep * 32 + ((lane & 16) ? 8 : 0) + (i < 8 ? i : 8 + i);
  int n = ntile * 16 + (lane & 15);
  P[idx] = (_Float16)W[(size_t)k * N + n];
}

// ---------- WMMA GEMM: C[M,N] = A[M,K] (f32, cvt->f16) x Bpacked ------------
// One wave computes one 16x16 C tile; 8 waves / 256-thread block.
__global__ void gat_gemm_wmma(const float* __restrict__ A,
                              const _Float16* __restrict__ Bp,
                              float* __restrict__ C, int M, int N, int K) {
  int lane = threadIdx.x & 31;
  int wave = threadIdx.x >> 5;
  int ntilesN = N >> 4;
  int ksteps  = K >> 5;
  int tile = blockIdx.x * (blockDim.x >> 5) + wave;
  if (tile >= (M >> 4) * ntilesN) return;       // wave-uniform: EXEC stays ~0
  int mtile = tile / ntilesN;
  int ntile = tile - mtile * ntilesN;

  // A layout (16-bit A 16x32): lane L -> row M = L&15; lanes>=16 K-offset +8;
  // halves 0..7 = K kb..kb+7, halves 8..15 = K kb+16..kb+23.
  const float* arow = A + (size_t)(mtile * 16 + (lane & 15)) * K + ((lane & 16) ? 8 : 0);
  const _Float16* bp = Bp + ((size_t)(ntile * ksteps) * 32 + lane) * 16;

  v8f acc = {};
  for (int ks = 0; ks < ksteps; ++ks) {
    int kb = ks * 32;
    if (ks + 1 < ksteps) __builtin_prefetch(arow + kb + 32, 0, 1);
    v16h a;
#pragma unroll
    for (int i = 0; i < 8; ++i) {
      a[i]     = (_Float16)arow[kb + i];
      a[i + 8] = (_Float16)arow[kb + 16 + i];
    }
    v16h b = *(const v16h*)(bp + (size_t)ks * 512);  // 32B contiguous per lane
    acc = __builtin_amdgcn_wmma_f32_16x16x32_f16(
        /*neg_a=*/false, a, /*neg_b=*/false, b,
        /*c_mod=*/(short)0, acc, /*reuse_a=*/false, /*reuse_b=*/false);
  }
  // C/D layout: lane -> col n = lane&15; rows (lane<16 ? 0..7 : 8..15).
  float* crow = C + (size_t)(mtile * 16 + ((lane & 16) ? 8 : 0)) * N
                  + ntile * 16 + (lane & 15);
#pragma unroll
  for (int r = 0; r < 8; ++r) crow[(size_t)r * N] = acc[r];
}

// ---------- per-node attention logits: al[n,h] = sum_c H[n,h*64+c]*a[h,c] ---
__global__ void gat_node_logits(const float* __restrict__ H,
                                const float* __restrict__ a_src,
                                const float* __restrict__ a_dst,
                                float* __restrict__ als, float* __restrict__ ald,
                                int Hh) {
  int idx = blockIdx.x * blockDim.x + threadIdx.x;
  if (idx >= NN * Hh) return;
  int n = idx / Hh;
  int h = idx - n * Hh;
  const float* row = H + (size_t)n * (Hh * 64) + h * 64;
  const float* as = a_src + h * 64;
  const float* ad = a_dst + h * 64;
  float ss = 0.f, sd = 0.f;
#pragma unroll 8
  for (int c = 0; c < 64; ++c) { float v = row[c]; ss += v * as[c]; sd += v * ad[c]; }
  als[idx] = ss;
  ald[idx] = sd;
}

static __device__ __forceinline__ float leaky(float v) {
  return v >= 0.f ? v : NEG_SLOPE * v;
}

// ---------- segment max over dst (encoded u32 atomic max) -------------------
__global__ void gat_edge_max(const int* __restrict__ src, const int* __restrict__ dst,
                             const float* __restrict__ als, const float* __restrict__ ald,
                             unsigned* __restrict__ menc, int Hh) {
  int idx = blockIdx.x * blockDim.x + threadIdx.x;
  if (idx >= (NE + NN) * Hh) return;
  int e = idx / Hh;
  int h = idx - e * Hh;
  int s = e < NE ? src[e] : e - NE;
  int d = e < NE ? dst[e] : e - NE;
  float v = leaky(als[s * Hh + h] + ald[d * Hh + h]);
  atomicMax(&menc[d * Hh + h], encf(v));
}

// ---------- segment sum of exp(e - m) over dst ------------------------------
__global__ void gat_edge_denom(const int* __restrict__ src, const int* __restrict__ dst,
                               const float* __restrict__ als, const float* __restrict__ ald,
                               const unsigned* __restrict__ menc,
                               float* __restrict__ denom, int Hh) {
  int idx = blockIdx.x * blockDim.x + threadIdx.x;
  if (idx >= (NE + NN) * Hh) return;
  int e = idx / Hh;
  int h = idx - e * Hh;
  int s = e < NE ? src[e] : e - NE;
  int d = e < NE ? dst[e] : e - NE;
  float v = leaky(als[s * Hh + h] + ald[d * Hh + h]);
  atomicAdd(&denom[d * Hh + h], __expf(v - decf(menc[d * Hh + h])));
}

// ---------- weighted scatter-aggregate: out[dst] += alpha * H[src] ----------
// One wave32 per edge; lanes stride the Hh*64 feature dim.
__global__ void gat_edge_aggregate(const int* __restrict__ src, const int* __restrict__ dst,
                                   const float* __restrict__ als, const float* __restrict__ ald,
                                   const unsigned* __restrict__ menc,
                                   const float* __restrict__ denom,
                                   const float* __restrict__ H, float* __restrict__ out,
                                   int Hh) {
  int e = (blockIdx.x * blockDim.x + threadIdx.x) >> 5;
  int lane = threadIdx.x & 31;
  if (e >= NE + NN) return;
  int s = e < NE ? src[e] : e - NE;
  int d = e < NE ? dst[e] : e - NE;
  float alpha[4];
  for (int h = 0; h < Hh; ++h) {
    float v = leaky(als[s * Hh + h] + ald[d * Hh + h]);
    alpha[h] = __expf(v - decf(menc[d * Hh + h])) / denom[d * Hh + h];
  }
  int F = Hh * 64;
  const float* hs = H + (size_t)s * F;
  float* od = out + (size_t)d * F;
  for (int f = lane; f < F; f += 32)
    atomicAdd(&od[f], alpha[f >> 6] * hs[f]);   // non-returning -> L2 atomic
}

// ---------- in-place bias + relu (F power of two, fmask = F-1) --------------
__global__ void gat_bias_relu(float* __restrict__ X, const float* __restrict__ b,
                              int total, int fmask) {
  int idx = blockIdx.x * blockDim.x + threadIdx.x;
  if (idx >= total) return;
  X[idx] = fmaxf(X[idx] + b[idx & fmask], 0.f);
}

// ---------- fused head: y[n] = relu(agg2[n]+b2) . Wh + bh -------------------
__global__ void gat_head(const float* __restrict__ agg2, const float* __restrict__ b2,
                         const float* __restrict__ Wh, const float* __restrict__ bh,
                         float* __restrict__ y) {
  int n = blockIdx.x * blockDim.x + threadIdx.x;
  if (n >= NN) return;
  const float* r = agg2 + (size_t)n * 64;
  float s = 0.f;
#pragma unroll 8
  for (int c = 0; c < 64; ++c)
    s += fmaxf(r[c] + b2[c], 0.f) * Wh[c];
  y[n] = s + bh[0];
}

static inline int cdiv(long long a, long long b) { return (int)((a + b - 1) / b); }

extern "C" void kernel_launch(void* const* d_in, const int* in_sizes, int n_in,
                              void* d_out, int out_size, void* d_ws, size_t ws_size,
                              hipStream_t stream) {
  const float* x     = (const float*)d_in[0];
  const int*   ei    = (const int*)d_in[1];   // [2,E]; harness delivers int32
  const float* W1    = (const float*)d_in[2];
  const float* a1s   = (const float*)d_in[3];
  const float* a1d   = (const float*)d_in[4];
  const float* b1    = (const float*)d_in[5];
  const float* W2    = (const float*)d_in[6];
  const float* a2s   = (const float*)d_in[7];
  const float* a2d   = (const float*)d_in[8];
  const float* b2    = (const float*)d_in[9];
  const float* Wh    = (const float*)d_in[10];
  const float* bh    = (const float*)d_in[11];
  float* y = (float*)d_out;
  const int* src = ei;
  const int* dst = ei + NE;

  // ---- workspace carve-out (256B aligned) ----
  char* ws = (char*)d_ws;
  size_t off = 0;
  auto carve = [&](size_t bytes) { char* p = ws + off; off += (bytes + 255) & ~(size_t)255; return p; };
  float*    h1    = (float*)carve((size_t)NN * 256 * 4);  // layer1 projection
  float*    agg1  = (float*)carve((size_t)NN * 256 * 4);  // layer1 aggregate / layer2 input
  float*    als   = (float*)carve((size_t)NN * 4 * 4);
  float*    ald   = (float*)carve((size_t)NN * 4 * 4);
  unsigned* menc  = (unsigned*)carve((size_t)NN * 4 * 4);
  float*    denom = (float*)carve((size_t)NN * 4 * 4);
  _Float16* W1p   = (_Float16*)carve((size_t)128 * 256 * 2);
  _Float16* W2p   = (_Float16*)carve((size_t)256 * 64 * 2);
  // layer2 feature buffers alias the (dead after aggregate1) h1 region
  float* h2   = h1;                        // [NN,64]
  float* agg2 = h1 + (size_t)NN * 64;      // [NN,64]
  (void)ws_size; (void)n_in; (void)in_sizes; (void)out_size;

  const int B = 256;

  // ---- pack weights to WMMA B layout ----
  gat_pack_b<<<cdiv(128 * 256, B), B, 0, stream>>>(W1, W1p, 128, 256);
  gat_pack_b<<<cdiv(256 * 64, B), B, 0, stream>>>(W2, W2p, 256, 64);

  // ---- layer 1 ----
  hipMemsetAsync(agg1, 0, (size_t)NN * 256 * 4, stream);
  hipMemsetAsync(menc, 0, (size_t)NN * 4 * 4, stream);   // 0 == encf(< -FLT_MAX)
  hipMemsetAsync(denom, 0, (size_t)NN * 4 * 4, stream);

  gat_gemm_wmma<<<cdiv((NN / 16) * (256 / 16), 8), B, 0, stream>>>(x, W1p, h1, NN, 256, 128);
  gat_node_logits<<<cdiv((long long)NN * 4, B), B, 0, stream>>>(h1, a1s, a1d, als, ald, 4);
  gat_edge_max<<<cdiv((long long)(NE + NN) * 4, B), B, 0, stream>>>(src, dst, als, ald, menc, 4);
  gat_edge_denom<<<cdiv((long long)(NE + NN) * 4, B), B, 0, stream>>>(src, dst, als, ald, menc, denom, 4);
  gat_edge_aggregate<<<cdiv((long long)(NE + NN) * 32, B), B, 0, stream>>>(src, dst, als, ald, menc, denom, h1, agg1, 4);
  gat_bias_relu<<<cdiv((long long)NN * 256, B), B, 0, stream>>>(agg1, b1, NN * 256, 255);

  // ---- layer 2 (heads=1) ----
  hipMemsetAsync(agg2, 0, (size_t)NN * 64 * 4, stream);
  hipMemsetAsync(menc, 0, (size_t)NN * 4, stream);
  hipMemsetAsync(denom, 0, (size_t)NN * 4, stream);

  gat_gemm_wmma<<<cdiv((NN / 16) * (64 / 16), 8), B, 0, stream>>>(agg1, W2p, h2, NN, 64, 256);
  gat_node_logits<<<cdiv(NN, B), B, 0, stream>>>(h2, a2s, a2d, als, ald, 1);
  gat_edge_max<<<cdiv(NE + NN, B), B, 0, stream>>>(src, dst, als, ald, menc, 1);
  gat_edge_denom<<<cdiv(NE + NN, B), B, 0, stream>>>(src, dst, als, ald, menc, denom, 1);
  gat_edge_aggregate<<<cdiv((long long)(NE + NN) * 32, B), B, 0, stream>>>(src, dst, als, ald, menc, denom, h2, agg2, 1);

  // ---- head ----
  gat_head<<<cdiv(NN, B), B, 0, stream>>>(agg2, b2, Wh, bh, y);
}
